// TransformerModel_49941879717922
// MI455X (gfx1250) — matrix operand order repaired and checked by reading, not compiled, not run on hardware
//
#include <hip/hip_runtime.h>
#include <hip/hip_bf16.h>
#include <math.h>

// ---------------- model constants ----------------
#define B_   4
#define SIN_ 1028
#define DIN_ 64
#define D_   1024
#define H_   16
#define HD_  64
#define S_   1024          // sequence length after two VALID k=3 convs
#define M_   (B_ * S_)     // 4096 token rows
#define EPS_ 1e-6f

typedef _Float16 v16h __attribute__((ext_vector_type(16)));
typedef _Float16 v8h  __attribute__((ext_vector_type(8)));
typedef float    v8f  __attribute__((ext_vector_type(8)));

// ---------------- optional CDNA5 async global->LDS DMA (ASYNCcnt path) ----------------
#if defined(__has_builtin)
# if __has_builtin(__builtin_amdgcn_global_load_async_to_lds_b128)
#  define USE_ASYNC_G2L 1
# endif
#endif
#ifndef USE_ASYNC_G2L
# define USE_ASYNC_G2L 0
#endif

#if USE_ASYNC_G2L
// signature (from clang diagnostic): (v4i32 as1*, v4i32 as3*, int offset, int cpol)
typedef int v4i_vs __attribute__((vector_size(4 * sizeof(int))));
typedef __attribute__((address_space(1))) v4i_vs g1_v4i;
typedef __attribute__((address_space(3))) v4i_vs l3_v4i;
static __device__ __forceinline__ void g2l_b128(const void* g, void* l) {
  __builtin_amdgcn_global_load_async_to_lds_b128((g1_v4i*)g, (l3_v4i*)l, 0, 0);
}
static __device__ __forceinline__ void wait_async_all() {
# if defined(__has_builtin) && __has_builtin(__builtin_amdgcn_s_wait_asynccnt)
  __builtin_amdgcn_s_wait_asynccnt(0);
# else
  asm volatile("s_wait_asynccnt 0" ::: "memory");
# endif
}
#endif

// ---------------- WMMA helpers (CDNA5 16x16x32 f16 -> f32) ----------------
static __device__ __forceinline__ v8f wmma_f16(v16h a, v16h b, v8f c) {
  return __builtin_amdgcn_wmma_f32_16x16x32_f16(
      false, a, false, b, (short)0, c, false, false);
}

// A fragment per CDNA5 16-bit A layout (lane<16: K {kb..kb+7, kb+16..kb+23})
static __device__ __forceinline__ v16h ld_a_frag(const _Float16* lds, int rowstride,
                                                 int row0, int kbase, int lane) {
  int m   = lane & 15;
  int sel = lane >> 4;
  const _Float16* base = lds + (row0 + m) * rowstride + kbase + sel * 8;
  union { v16h v; v8h h[2]; } u;
  u.h[0] = *(const v8h*)(base);
  u.h[1] = *(const v8h*)(base + 16);
  return u.v;
}

// B fragment: lane l = column N=col0+(l&15), K {8*(l>>4)..+7, 16+8*(l>>4)..+7}
static __device__ __forceinline__ v16h ld_b_frag(const _Float16* lds, int rowstride,
                                                 int row0, int col0, int lane) {
  const _Float16* base = lds + (row0 + (lane >> 4) * 8) * rowstride + col0 + (lane & 15);
  v16h r;
#pragma unroll
  for (int s = 0; s < 8; ++s) {
    r[s]     = base[s * rowstride];
    r[s + 8] = base[(s + 16) * rowstride];
  }
  return r;
}

// ---------------- conv1 (small: 0.8 GF, VALU) -> f16 activations ----------------
__global__ void conv1_kernel(const float* __restrict__ x, const float* __restrict__ w,
                             const float* __restrict__ bias, _Float16* __restrict__ h1) {
  int idx = blockIdx.x * blockDim.x + threadIdx.x;       // B * 1026 * 512
  int c = idx & 511;
  int s = (idx >> 9) % 1026;
  int b = idx / (512 * 1026);
  float acc = bias[c];
  const float* wc = w + c * (DIN_ * 3);
  for (int t = 0; t < 3; ++t) {
    const float* xr = x + ((size_t)(b * SIN_ + s + t)) * DIN_;
    const float* wr = wc + t;
#pragma unroll 8
    for (int ic = 0; ic < DIN_; ++ic) acc = fmaf(xr[ic], wr[ic * 3], acc);
  }
  h1[((size_t)(b * 1026 + s)) * 512 + c] = (_Float16)acc;
}

// ---------------- positional encoding add ----------------
__global__ void pe_add_kernel(float* __restrict__ h) {
  int idx = blockIdx.x * blockDim.x + threadIdx.x;       // B * 1024 * 1024
  int c = idx & 1023;
  int s = (idx >> 10) & 1023;
  float freq = __expf(-(float)(c & ~1) * (9.210340371976184f / (float)D_));
  float ang  = (float)s * freq;
  h[idx] += (c & 1) ? __cosf(ang) : __sinf(ang);
}

// ---------------- layernorm (torch-style, ddof=1, eps on std), f32 -> f16 ----------------
__global__ void layernorm_kernel(const float* __restrict__ x, const float* __restrict__ w,
                                 const float* __restrict__ bias, _Float16* __restrict__ out) {
  __shared__ float red[256];
  int row = blockIdx.x;
  int tid = threadIdx.x;
  const float* xr = x + (size_t)row * D_;
  float v[4];
  float sum = 0.f;
#pragma unroll
  for (int j = 0; j < 4; ++j) { v[j] = xr[tid + j * 256]; sum += v[j]; }
  red[tid] = sum; __syncthreads();
  for (int s = 128; s > 0; s >>= 1) { if (tid < s) red[tid] += red[tid + s]; __syncthreads(); }
  float mean = red[0] * (1.0f / D_);
  __syncthreads();
  float ss = 0.f;
#pragma unroll
  for (int j = 0; j < 4; ++j) { float d = v[j] - mean; ss += d * d; }
  red[tid] = ss; __syncthreads();
  for (int s = 128; s > 0; s >>= 1) { if (tid < s) red[tid] += red[tid + s]; __syncthreads(); }
  float var = red[0] * (1.0f / (D_ - 1));
  float inv = __builtin_amdgcn_rcpf(sqrtf(var) + EPS_);   // v_rcp_f32, no div sequence
#pragma unroll
  for (int j = 0; j < 4; ++j) {
    int d = tid + j * 256;
    out[(size_t)row * D_ + d] = (_Float16)(w[d] * ((v[j] - mean) * inv) + bias[d]);
  }
}

// ---------------- tiled WMMA GEMM, ping-pong double buffered ----------------
// C[M,N] = act(A[M,K]_f16 @ W[N,K]^T [+ bias]) (+res) [*0.125]
// block 128x128, K-step 32; SKEW rows added per 1024-row batch (conv2-as-GEMM),
// WSTRIDE: element stride along K in W (3 for torch Conv1d [out][in][k] layout).
template<bool RELU, bool RES, bool OUTF16, bool BIAS, bool SCALE8, int WSTRIDE>
__global__ void gemm_wmma(const _Float16* __restrict__ A, const float* __restrict__ W,
                          const float* __restrict__ bias, void* __restrict__ Cout,
                          int N, int K, int skew, int toff) {
  constexpr int SAS = 40;    // padded A row stride (halves): breaks 64-bank periodicity
  constexpr int SWS = 136;   // padded W row stride (halves)
  __shared__ _Float16 sA[2][128 * SAS];   // [row][k]
  __shared__ _Float16 sW[2][32 * SWS];    // [k][n] transposed weight tile
  const int tid  = threadIdx.x;
  const int lane = tid & 31;
  const int wave = tid >> 5;
  const int wm   = wave >> 1;
  const int wn   = wave & 1;
  const int n0   = blockIdx.x * 128;
  const int m0   = blockIdx.y * 128;
  const int ar = tid >> 1, ak = (tid & 1) * 16;
  const int wr = tid >> 1, wk = (tid & 1) * 16;

  // conv2-as-GEMM: blocks never straddle batches (1024 % 128 == 0)
  const _Float16* Aeff = A + (size_t)((m0 >> 10) * skew + toff) * K;

  v8f acc[2][4];
#pragma unroll
  for (int i = 0; i < 2; ++i)
#pragma unroll
    for (int j = 0; j < 4; ++j) acc[i][j] = (v8f)(0.0f);

  float wt[16];
  auto load_w = [&](int kk) {
    const float* src = W + (size_t)(n0 + wr) * K * WSTRIDE + (size_t)(kk + wk) * WSTRIDE;
    if constexpr (WSTRIDE == 1) {
      float4 f0 = *(const float4*)(src);
      float4 f1 = *(const float4*)(src + 4);
      float4 f2 = *(const float4*)(src + 8);
      float4 f3 = *(const float4*)(src + 12);
      wt[0]=f0.x; wt[1]=f0.y; wt[2]=f0.z; wt[3]=f0.w;
      wt[4]=f1.x; wt[5]=f1.y; wt[6]=f1.z; wt[7]=f1.w;
      wt[8]=f2.x; wt[9]=f2.y; wt[10]=f2.z; wt[11]=f2.w;
      wt[12]=f3.x; wt[13]=f3.y; wt[14]=f3.z; wt[15]=f3.w;
    } else {
#pragma unroll
      for (int j = 0; j < 16; ++j) wt[j] = src[j * WSTRIDE];
    }
  };
  auto store_w = [&](int buf) {
#pragma unroll
    for (int j = 0; j < 16; ++j) sW[buf][(wk + j) * SWS + wr] = (_Float16)wt[j];
  };
#if USE_ASYNC_G2L
  auto stage_a = [&](int buf, int kk) {   // DMA: no VGPR round trip, ASYNCcnt tracked
    const _Float16* src = Aeff + (size_t)(m0 + ar) * K + kk + ak;
    _Float16* dst = &sA[buf][ar * SAS + ak];
    g2l_b128(src, dst);
    g2l_b128(src + 8, dst + 8);
  };
#else
  v8h a0, a1;
  auto load_a = [&](int kk) {
    const _Float16* src = Aeff + (size_t)(m0 + ar) * K + kk + ak;
    a0 = *(const v8h*)(src);
    a1 = *(const v8h*)(src + 8);
  };
  auto store_a = [&](int buf) {
    *(v8h*)(&sA[buf][ar * SAS + ak])     = a0;
    *(v8h*)(&sA[buf][ar * SAS + ak + 8]) = a1;
  };
#endif

  // prologue: fill buffer 0
  load_w(0);
#if USE_ASYNC_G2L
  stage_a(0, 0);
#else
  load_a(0);
  store_a(0);
#endif
  store_w(0);
#if USE_ASYNC_G2L
  wait_async_all();
#endif
  __syncthreads();

  int buf = 0;
  for (int kk = 0; kk < K; kk += 32) {
    const bool nx = (kk + 32) < K;
    if (nx) {               // issue next tile's global traffic before the WMMA block
#if USE_ASYNC_G2L
      stage_a(buf ^ 1, kk + 32);
#else
      load_a(kk + 32);
#endif
      load_w(kk + 32);
    }
    // ---- compute on current buffer ----
    v16h af0 = ld_a_frag(sA[buf], SAS, wm * 32,      0, lane);
    v16h af1 = ld_a_frag(sA[buf], SAS, wm * 32 + 16, 0, lane);
#pragma unroll
    for (int j = 0; j < 4; ++j) {
      v16h bf = ld_b_frag(sW[buf], SWS, 0, wn * 64 + j * 16, lane);
      acc[0][j] = wmma_f16(af0, bf, acc[0][j]);
      acc[1][j] = wmma_f16(af1, bf, acc[1][j]);
    }
    if (nx) {
      store_w(buf ^ 1);
#if !USE_ASYNC_G2L
      store_a(buf ^ 1);
#else
      wait_async_all();
#endif
    }
    __syncthreads();
    buf ^= 1;
  }

  // epilogue: C frag layout (row = p + 8*(lane>>4), col = lane&15)
  float*    Cf = (float*)Cout;
  _Float16* Ch = (_Float16*)Cout;
  const int rsel = (lane >> 4) * 8;
  const int nlo  = lane & 15;
#pragma unroll
  for (int i = 0; i < 2; ++i) {
#pragma unroll
    for (int j = 0; j < 4; ++j) {
      int gc = n0 + wn * 64 + j * 16 + nlo;
      float bi = BIAS ? bias[gc] : 0.0f;
#pragma unroll
      for (int p = 0; p < 8; ++p) {
        int gr = m0 + wm * 32 + i * 16 + rsel + p;
        float v = acc[i][j][p] + bi;
        if (RELU)   v = fmaxf(v, 0.0f);
        if (SCALE8) v *= 0.125f;          // fold attention 1/sqrt(HD) into Q proj
        size_t o = (size_t)gr * N + gc;
        if (RES) v += Cf[o];
        if (OUTF16) Ch[o] = (_Float16)v; else Cf[o] = v;
      }
    }
  }
}

// ---------------- fused softsign attention (flash-style) ----------------
// grid: (S/128, B*H); block 256; per block 128 query rows of one (b,h)
// Q is pre-scaled by 1/sqrt(HD) in its projection epilogue.
__global__ void attn_kernel(const _Float16* __restrict__ q, const _Float16* __restrict__ k,
                            const _Float16* __restrict__ v, _Float16* __restrict__ wv) {
  constexpr int ST = 72;              // padded row stride (halves)
  __shared__ _Float16 sQ[128 * ST];   // [m][d]
  __shared__ _Float16 sK[64 * ST];    // [d][t] transposed
  __shared__ _Float16 sV[64 * ST];    // [t][d]
  __shared__ _Float16 sAt[128 * ST];  // [m][t] softsigned scores

  const int tid  = threadIdx.x;
  const int lane = tid & 31;
  const int wave = tid >> 5;
  const int m0   = wave * 16;
  const int s0   = blockIdx.x * 128;
  const int b    = blockIdx.y >> 4;
  const int hh   = blockIdx.y & 15;
  const size_t brow = (size_t)b * S_;
  const int ch   = hh * HD_;

  // stage Q tile once
  {
    int row = tid >> 1, off = (tid & 1) * 32;
    const _Float16* src = q + (brow + s0 + row) * D_ + ch + off;
    _Float16* dst = sQ + row * ST + off;
#if USE_ASYNC_G2L
#pragma unroll
    for (int j = 0; j < 4; ++j) g2l_b128(src + j * 8, dst + j * 8);
    wait_async_all();
#else
#pragma unroll
    for (int j = 0; j < 4; ++j) *(v8h*)(dst + j * 8) = *(const v8h*)(src + j * 8);
#endif
  }
  __syncthreads();

  v16h aq[2];
#pragma unroll
  for (int ks = 0; ks < 2; ++ks) aq[ks] = ld_a_frag(sQ, ST, m0, ks * 32, lane);

  v8f wvacc[4];
#pragma unroll
  for (int j = 0; j < 4; ++j) wvacc[j] = (v8f)(0.0f);

  const int rsel = (lane >> 4) * 8;
  const int nlo  = lane & 15;

  for (int t0 = 0; t0 < S_; t0 += 64) {
    // V tile straight copy (DMA if available)
    {
      int row = tid >> 2, part = (tid & 3) * 16;
      const _Float16* src = v + (brow + t0 + row) * D_ + ch + part;
      _Float16* dst = sV + row * ST + part;
#if USE_ASYNC_G2L
      g2l_b128(src, dst);
      g2l_b128(src + 8, dst + 8);
#else
      *(v8h*)(dst)     = *(const v8h*)(src);
      *(v8h*)(dst + 8) = *(const v8h*)(src + 8);
#endif
    }
    // K tile transposed (needs per-element scatter)
    {
      int tcol = tid & 63, dh = tid >> 6;
      const _Float16* src = k + (brow + t0 + tcol) * D_ + ch + dh * 16;
      v8h k0 = *(const v8h*)(src);
      v8h k1 = *(const v8h*)(src + 8);
#pragma unroll
      for (int j = 0; j < 8; ++j) sK[(dh * 16 + j) * ST + tcol] = k0[j];
#pragma unroll
      for (int j = 0; j < 8; ++j) sK[(dh * 16 + 8 + j) * ST + tcol] = k1[j];
    }
#if USE_ASYNC_G2L
    wait_async_all();
#endif
    __syncthreads();

    // scores = softsign(Qs K^T) -> sAt (per-wave 16-row slab)
#pragma unroll
    for (int j = 0; j < 4; ++j) {
      v8f sc = (v8f)(0.0f);
#pragma unroll
      for (int ks = 0; ks < 2; ++ks) {
        v16h bk = ld_b_frag(sK, ST, ks * 32, j * 16, lane);
        sc = wmma_f16(aq[ks], bk, sc);
      }
#pragma unroll
      for (int p = 0; p < 8; ++p) {
        float s = sc[p];
        s = s * __builtin_amdgcn_rcpf(1.0f + fabsf(s));   // fast softsign: v_rcp_f32
        sAt[(m0 + rsel + p) * ST + j * 16 + nlo] = (_Float16)s;
      }
    }
    // wv += attn @ V  (same-wave LDS round trip re-layouts C->A fragments)
#pragma unroll
    for (int j = 0; j < 4; ++j) {
#pragma unroll
      for (int ks = 0; ks < 2; ++ks) {
        v16h aa = ld_a_frag(sAt, ST, m0, ks * 32, lane);
        v16h bv = ld_b_frag(sV, ST, ks * 32, j * 16, lane);
        wvacc[j] = wmma_f16(aa, bv, wvacc[j]);
      }
    }
    __syncthreads();
  }

#pragma unroll
  for (int j = 0; j < 4; ++j) {
#pragma unroll
    for (int p = 0; p < 8; ++p) {
      int gr = s0 + m0 + rsel + p;
      int gc = ch + j * 16 + nlo;
      wv[(brow + gr) * D_ + gc] = (_Float16)wvacc[j][p];
    }
  }
}

// ---------------- mean-pool over S + final FC ----------------
__global__ void pool_fc_kernel(const float* __restrict__ h, const float* __restrict__ fcw,
                               const float* __restrict__ fcb, float* __restrict__ out) {
  __shared__ float red[256];
  int b = blockIdx.x, tid = threadIdx.x;
  float partial = 0.f;
  for (int d = tid; d < D_; d += 256) {
    float s = 0.f;
    const float* col = h + ((size_t)b * S_) * D_ + d;
    for (int srow = 0; srow < S_; ++srow) s += col[(size_t)srow * D_];
    partial += (s * (1.0f / S_)) * fcw[d];
  }
  red[tid] = partial; __syncthreads();
  for (int s = 128; s > 0; s >>= 1) { if (tid < s) red[tid] += red[tid + s]; __syncthreads(); }
  if (tid == 0) out[b] = red[0] + fcb[0];
}

// ---------------- launch ----------------
extern "C" void kernel_launch(void* const* d_in, const int* in_sizes, int n_in,
                              void* d_out, int out_size, void* d_ws, size_t ws_size,
                              hipStream_t stream) {
  const float* x       = (const float*)d_in[0];
  const float* conv1_w = (const float*)d_in[1];
  const float* conv1_b = (const float*)d_in[2];
  const float* conv2_w = (const float*)d_in[3];
  const float* conv2_b = (const float*)d_in[4];
  const float* lnA_w   = (const float*)d_in[5];
  const float* lnA_b   = (const float*)d_in[6];
  const float* q_w     = (const float*)d_in[7];
  const float* q_b     = (const float*)d_in[8];
  const float* k_w     = (const float*)d_in[9];
  const float* k_b     = (const float*)d_in[10];
  const float* v_w     = (const float*)d_in[11];
  const float* v_b     = (const float*)d_in[12];
  const float* o_w     = (const float*)d_in[13];
  const float* o_b     = (const float*)d_in[14];
  const float* lnB_w   = (const float*)d_in[15];
  const float* lnB_b   = (const float*)d_in[16];
  const float* f1_w    = (const float*)d_in[17];
  const float* f1_b    = (const float*)d_in[18];
  const float* f2_w    = (const float*)d_in[19];
  const float* f2_b    = (const float*)d_in[20];
  const float* fc_w    = (const float*)d_in[21];
  const float* fc_b    = (const float*)d_in[22];
  float* out = (float*)d_out;

  char* ws = (char*)d_ws;
  const size_t OFF_H  = 0;                                    // f32 [4096][1024]
  const size_t OFF_H1 = OFF_H  + (size_t)M_ * D_ * 4;         // f16 [4][1026][512]
  const size_t OFF_HN = OFF_H1 + (size_t)B_ * 1026 * 512 * 2;
  const size_t OFF_Q  = OFF_HN + (size_t)M_ * D_ * 2;
  const size_t OFF_K  = OFF_Q  + (size_t)M_ * D_ * 2;
  const size_t OFF_V  = OFF_K  + (size_t)M_ * D_ * 2;
  const size_t OFF_WV = OFF_V  + (size_t)M_ * D_ * 2;
  const size_t OFF_FF = OFF_WV + (size_t)M_ * D_ * 2;         // f16 [4096][2048]
  float*    h   = (float*)(ws + OFF_H);
  _Float16* h1  = (_Float16*)(ws + OFF_H1);
  _Float16* hn  = (_Float16*)(ws + OFF_HN);
  _Float16* qb  = (_Float16*)(ws + OFF_Q);
  _Float16* kb  = (_Float16*)(ws + OFF_K);
  _Float16* vb  = (_Float16*)(ws + OFF_V);
  _Float16* wvb = (_Float16*)(ws + OFF_WV);
  _Float16* ffb = (_Float16*)(ws + OFF_FF);

  const dim3 gN1(D_ / 128, M_ / 128);
  const dim3 gN2((2 * D_) / 128, M_ / 128);
  const dim3 gAt(S_ / 128, B_ * H_);

  // conv frontend: conv1 VALU -> f16; conv2 as 3 shifted WMMA GEMMs (K=512, W stride 3)
  conv1_kernel<<<(B_ * 1026 * 512) / 256, 256, 0, stream>>>(x, conv1_w, conv1_b, h1);
  gemm_wmma<false, false, false, true,  false, 3><<<gN1, 256, 0, stream>>>(
      h1, conv2_w + 0, conv2_b, h, D_, 512, /*skew=*/2, /*toff=*/0);
  gemm_wmma<false, true,  false, false, false, 3><<<gN1, 256, 0, stream>>>(
      h1, conv2_w + 1, conv2_b, h, D_, 512, 2, 1);
  gemm_wmma<false, true,  false, false, false, 3><<<gN1, 256, 0, stream>>>(
      h1, conv2_w + 2, conv2_b, h, D_, 512, 2, 2);
  pe_add_kernel<<<(B_ * S_ * D_) / 256, 256, 0, stream>>>(h);

  for (int l = 0; l < 4; ++l) {
    const size_t dd = (size_t)l * D_ * D_;
    const size_t d2 = (size_t)l * 2 * D_ * D_;
    layernorm_kernel<<<M_, 256, 0, stream>>>(h, lnA_w + l * D_, lnA_b + l * D_, hn);
    // Q pre-scaled by 0.125 = 1/sqrt(HD)
    gemm_wmma<false, false, true, true, true,  1><<<gN1, 256, 0, stream>>>(
        hn, q_w + dd, q_b + l * D_, qb, D_, D_, 0, 0);
    gemm_wmma<false, false, true, true, false, 1><<<gN1, 256, 0, stream>>>(
        hn, k_w + dd, k_b + l * D_, kb, D_, D_, 0, 0);
    gemm_wmma<false, false, true, true, false, 1><<<gN1, 256, 0, stream>>>(
        hn, v_w + dd, v_b + l * D_, vb, D_, D_, 0, 0);
    attn_kernel<<<gAt, 256, 0, stream>>>(qb, kb, vb, wvb);
    gemm_wmma<false, true, false, true, false, 1><<<gN1, 256, 0, stream>>>(
        wvb, o_w + dd, o_b + l * D_, h, D_, D_, 0, 0);
    layernorm_kernel<<<M_, 256, 0, stream>>>(h, lnB_w + l * D_, lnB_b + l * D_, hn);
    gemm_wmma<true,  false, true, true, false, 1><<<gN2, 256, 0, stream>>>(
        hn,  f1_w + d2, f1_b + l * 2 * D_, ffb, 2 * D_, D_, 0, 0);
    gemm_wmma<false, true, false, true, false, 1><<<gN1, 256, 0, stream>>>(
        ffb, f2_w + d2, f2_b + l * D_,     h,   D_, 2 * D_, 0, 0);
  }

  pool_fc_kernel<<<B_, 256, 0, stream>>>(h, fc_w, fc_b, out);
}
